// GPT_45930380264310
// MI455X (gfx1250) — matrix operand order repaired and checked by reading, not compiled, not run on hardware
//
#include <hip/hip_runtime.h>
#include <hip/hip_bf16.h>

#define LL 4
#define BB 2
#define TT 2048
#define CC 512
#define HH 8
#define HDIM 64
#define VV 50257
#define BT (BB * TT)   // 4096

typedef __bf16 v16bf __attribute__((ext_vector_type(16)));
typedef __bf16 v8bf  __attribute__((ext_vector_type(8)));
typedef float  v8f   __attribute__((ext_vector_type(8)));

// fp32 -> bf16 round-to-nearest-even (producer/convert kernels only)
__device__ __forceinline__ __bf16 f2bf(float f) {
    unsigned u = __builtin_bit_cast(unsigned, f);
    u += 0x7FFFu + ((u >> 16) & 1u);
    unsigned short s = (unsigned short)(u >> 16);
    return __builtin_bit_cast(__bf16, s);
}

#define CAT16(lo, hi) __builtin_shufflevector(lo, hi, 0,1,2,3,4,5,6,7,8,9,10,11,12,13,14,15)

__device__ __forceinline__ float gelu_f(float x) {
    float x3 = x * x * x;
    return 0.5f * x * (1.0f + tanhf(0.7978845608028654f * (x + 0.044715f * x3)));
}

// Async copy of 16B global -> LDS (CDNA5 GLOBAL_LOAD_ASYNC_TO_LDS_B128).
// ldsoff = LDS byte offset (low 32 bits of generic shared address);
// INST_OFFSET applies to both global and LDS sides (ISA 15.18.3).
__device__ __forceinline__ void async_g2l_b128(unsigned ldsoff, unsigned long long gaddr) {
    asm volatile("global_load_async_to_lds_b128 %0, %1, off"
                 :: "v"(ldsoff), "v"(gaddr) : "memory");
}
__device__ __forceinline__ void async_g2l_b128_o16(unsigned ldsoff, unsigned long long gaddr) {
    asm volatile("global_load_async_to_lds_b128 %0, %1, off offset:16"
                 :: "v"(ldsoff), "v"(gaddr) : "memory");
}
__device__ __forceinline__ void wait_async0() {
    asm volatile("s_wait_asynccnt 0x0" ::: "memory");
}

// Swizzled A-tile layout for [M,K] bf16 activations: 16x32 tiles, lane-contiguous.
__device__ __forceinline__ size_t aswz(int m, int k, int K) {
    const int mt = m >> 4, lm = m & 15;
    const int kt = k >> 5, kk = k & 31;
    const int sub = kk >> 3;                       // 0..3
    const int half = sub & 1;
    const int e = (kk & 7) + ((sub >> 1) << 3);
    return (((size_t)mt * (K >> 5) + kt) << 9) + ((half << 4) + lm) * 16 + e;
}

// ---------------------------------------------------------------------------
// Embedding: x[b,t,:] = wte[idx[b,t],:] + wpe[t,:]   (fp32 residual stream)
// ---------------------------------------------------------------------------
__global__ __launch_bounds__(256)
void embed_kernel(const float* __restrict__ wte, const float* __restrict__ wpe,
                  const int* __restrict__ idx, float* __restrict__ x) {
    size_t i = (size_t)blockIdx.x * 256 + threadIdx.x;
    if (i >= (size_t)BT * CC) return;
    int c = (int)(i % CC);
    size_t bt = i / CC;
    int t = (int)(bt % TT);
    x[i] = wte[(size_t)idx[bt] * CC + c] + wpe[(size_t)t * CC + c];
}

// ---------------------------------------------------------------------------
// Weight transpose+convert:  W[K,N] fp32  ->  Wb[N,K] bf16   (K,N mult of 32)
// ---------------------------------------------------------------------------
__global__ __launch_bounds__(256)
void convtr_kernel(const float* __restrict__ W, __bf16* __restrict__ Wb, int K, int N) {
    __shared__ float tile[32][33];
    const int tx = threadIdx.x & 31;
    const int ty = threadIdx.x >> 5;     // 0..7
    const int k0 = blockIdx.y * 32;
    const int n0 = blockIdx.x * 32;
#pragma unroll
    for (int i = 0; i < 4; ++i)
        tile[ty + 8 * i][tx] = W[(size_t)(k0 + ty + 8 * i) * N + (n0 + tx)];
    __syncthreads();
#pragma unroll
    for (int i = 0; i < 4; ++i)
        Wb[(size_t)(n0 + ty + 8 * i) * K + (k0 + tx)] = f2bf(tile[tx][ty + 8 * i]);
}

// Straight convert (wte is already [N,K] for the tied head)
__global__ __launch_bounds__(256)
void conv_kernel(const float* __restrict__ W, __bf16* __restrict__ Wb, size_t n) {
    size_t i = (size_t)blockIdx.x * 256 + threadIdx.x;
    if (i < n) Wb[i] = f2bf(W[i]);
}

// ---------------------------------------------------------------------------
// LayerNorm over last dim (C=512); bf16 output in swizzled A-tile layout
// ---------------------------------------------------------------------------
__global__ __launch_bounds__(256)
void ln_kernel(const float* __restrict__ x, const float* __restrict__ sc,
               const float* __restrict__ bi, __bf16* __restrict__ out) {
    const int r = blockIdx.x, t = threadIdx.x;
    const float* xr = x + (size_t)r * CC;
    __shared__ float red[256];
    float ls = 0.f;
    for (int c = t; c < CC; c += 256) ls += xr[c];
    red[t] = ls; __syncthreads();
    for (int s = 128; s > 0; s >>= 1) { if (t < s) red[t] += red[t + s]; __syncthreads(); }
    float mean = red[0] / (float)CC;
    __syncthreads();
    float lv = 0.f;
    for (int c = t; c < CC; c += 256) { float d = xr[c] - mean; lv += d * d; }
    red[t] = lv; __syncthreads();
    for (int s = 128; s > 0; s >>= 1) { if (t < s) red[t] += red[t + s]; __syncthreads(); }
    float rstd = rsqrtf(red[0] / (float)CC + 1e-5f);
    for (int c = t; c < CC; c += 256)
        out[aswz(r, c, CC)] = f2bf((xr[c] - mean) * rstd * sc[c] + bi[c]);
}

// ---------------------------------------------------------------------------
// bf16 WMMA GEMM: Y[M,N] = act(X @ Wb^T + bias) (+ residual)
//   Xs: swizzled A-tile bf16 (direct v16bf loads).
//   Wb: [N,K] bf16; the 64x32 B tile (shared by all 4 waves of the block) is
//   staged into LDS via async global->LDS copies, double-buffered so the copy
//   of step k+1 overlaps the WMMAs of step k (ASYNCcnt + one barrier/step).
// Wave tile: 32 rows x 64 cols; 4 waves/block.
// OUTM: 0 = fp32 row-major, 1 = bf16 row-major, 2 = bf16 swizzled A-tile
// ---------------------------------------------------------------------------
template <bool GELU_ACT, bool RES, bool BIAS, int OUTM>
__global__ __launch_bounds__(128)
void wmma_gemm(const __bf16* __restrict__ Xs, const __bf16* __restrict__ Wb,
               const float* __restrict__ bias, const float* __restrict__ res,
               void* __restrict__ Yv, int M, int N, int K) {
    __shared__ __bf16 s_b[2][64][32];                 // [buf][n_local][k_local], 4KB each
    const int tid = threadIdx.x;
    const int lane = tid & 31;
    const int wave = tid >> 5;
    const int mt0 = (blockIdx.y * 4 + wave) * 2;      // two 16-row tiles
    const int nb = blockIdx.x * 64;
    const int lm = lane & 15, lh = lane >> 4;
    const int KT = K >> 5;

    const __bf16* a0p = Xs + (((size_t)(mt0 + 0) * KT) << 9) + lane * 16;
    const __bf16* a1p = Xs + (((size_t)(mt0 + 1) * KT) << 9) + lane * 16;

    // staging: thread tid moves 32B: row = tid>>1 (n_local), part = tid&1 (16 k's)
    const int srow = tid >> 1, spart = tid & 1;
    int gn = nb + srow; if (gn >= N) gn = N - 1;       // clamp: never stored
    const __bf16* gsrc = Wb + (size_t)gn * K + spart * 16;
    const unsigned lo0 = (unsigned)(unsigned long long)&s_b[0][srow][spart * 16];
    const unsigned lo1 = (unsigned)(unsigned long long)&s_b[1][srow][spart * 16];

    // preload k0 = 0 into buffer 0
    async_g2l_b128(lo0, (unsigned long long)gsrc);
    async_g2l_b128_o16(lo0, (unsigned long long)gsrc);

    v8f acc[2][4] = {};
    int buf = 0;
    for (int k0 = 0; k0 < K; k0 += 32) {
        wait_async0();
        __syncthreads();                               // current buffer ready; other buffer free
        if (k0 + 32 < K) {
            const unsigned long long g = (unsigned long long)(gsrc + k0 + 32);
            const unsigned lo = buf ? lo0 : lo1;
            async_g2l_b128(lo, g);
            async_g2l_b128_o16(lo, g);
        }
        v16bf a0 = *(const v16bf*)a0p; a0p += 512;
        v16bf a1 = *(const v16bf*)a1p; a1p += 512;
        v16bf b[4];
#pragma unroll
        for (int j = 0; j < 4; ++j)
            b[j] = CAT16(*(const v8bf*)(&s_b[buf][j * 16 + lm][lh * 8]), *(const v8bf*)(&s_b[buf][j * 16 + lm][16 + lh * 8]));
#pragma unroll
        for (int j = 0; j < 4; ++j)
            acc[0][j] = __builtin_amdgcn_wmma_f32_16x16x32_bf16(
                false, a0, false, b[j], (short)0, acc[0][j], false, false);
#pragma unroll
        for (int j = 0; j < 4; ++j)
            acc[1][j] = __builtin_amdgcn_wmma_f32_16x16x32_bf16(
                false, a1, false, b[j], (short)0, acc[1][j], false, false);
        buf ^= 1;
    }

    // ---- Epilogue ----
#pragma unroll
    for (int mi = 0; mi < 2; ++mi)
#pragma unroll
        for (int j = 0; j < 4; ++j) {
            const int n = nb + j * 16 + lm;
            if (n >= N) continue;
            const float bv = BIAS ? bias[n] : 0.f;
#pragma unroll
            for (int v = 0; v < 8; ++v) {
                const int m = (mt0 + mi) * 16 + v + 8 * lh;
                float val = acc[mi][j][v] + bv;
                if (GELU_ACT) val = gelu_f(val);
                if (RES) val += res[(size_t)m * N + n];
                if (OUTM == 0)      ((float*)Yv)[(size_t)m * N + n] = val;
                else if (OUTM == 1) ((__bf16*)Yv)[(size_t)m * N + n] = f2bf(val);
                else                ((__bf16*)Yv)[aswz(m, n, N)] = f2bf(val);
            }
        }
}

// ---------------------------------------------------------------------------
// Flash attention on bf16 qkv [B,T,3C] (row-major): one wave per (b,h,16-q tile).
// K-frags: contiguous v16bf loads. V: coalesced row loads -> LDS transpose ->
// contiguous v16bf LDS fragment loads. 1/sqrt(HD) folded into softmax.
// Output y: bf16 swizzled A-tile (feeds proj GEMM).
// ---------------------------------------------------------------------------
__global__ __launch_bounds__(32)
void attn_kernel(const __bf16* __restrict__ qkv, __bf16* __restrict__ y) {
    const int lane = threadIdx.x;
    const int lm = lane & 15, lh = lane >> 4;
    const int q0 = blockIdx.x * 16;
    const int hh = blockIdx.y;
    const int b  = blockIdx.z;
    const size_t row3C = 3 * (size_t)CC;

    __shared__ float  s_s[16][36];
    __shared__ __bf16 s_p[16][36];
    __shared__ __bf16 s_vt[64][48];     // V transposed: [d][key], 32B-aligned rows
    __shared__ float  s_max[16], s_sum[16], s_scale[16];

    // Q A-fragments (16x64 as two 16x32 chunks), raw bf16
    const v8bf* qrow = (const v8bf*)(qkv + ((size_t)b * TT + q0 + lm) * row3C + hh * HDIM);
    v16bf qa0 = CAT16(qrow[2 * lh],     qrow[2 * lh + 1]);
    v16bf qa1 = CAT16(qrow[4 + 2 * lh], qrow[5 + 2 * lh]);

    if (lane < 16) { s_max[lane] = -__builtin_inff(); s_sum[lane] = 0.f; }
    __syncthreads();

    v8f o[4] = {};
    const int qend = q0 + 15;

    for (int kb = 0; kb <= qend; kb += 32) {
        // ---- stage V block (32 keys x 64 dims) transposed into LDS ----
        {
            int key = kb + lane; if (key > TT - 1) key = TT - 1;
            const __bf16* vrow = qkv + ((size_t)b * TT + key) * row3C + 2 * CC + hh * HDIM;
            const v16bf r0 = *(const v16bf*)(vrow);
            const v16bf r1 = *(const v16bf*)(vrow + 16);
            const v16bf r2 = *(const v16bf*)(vrow + 32);
            const v16bf r3 = *(const v16bf*)(vrow + 48);
#pragma unroll
            for (int e = 0; e < 16; ++e) {
                s_vt[e][lane]      = r0[e];
                s_vt[16 + e][lane] = r1[e];
                s_vt[32 + e][lane] = r2[e];
                s_vt[48 + e][lane] = r3[e];
            }
        }
        // ---- S = Q K^T (16 x 32) ----
#pragma unroll
        for (int j = 0; j < 2; ++j) {
            int key = kb + j * 16 + lm; if (key > TT - 1) key = TT - 1;
            const __bf16* krow = qkv + ((size_t)b * TT + key) * row3C + CC + hh * HDIM;
            v16bf k0f = *(const v16bf*)(krow + lh * 16);
            v16bf k1f = *(const v16bf*)(krow + 32 + lh * 16);
            v8f s = {};
            s = __builtin_amdgcn_wmma_f32_16x16x32_bf16(false, qa0, false, k0f, (short)0, s, false, false);
            s = __builtin_amdgcn_wmma_f32_16x16x32_bf16(false, qa1, false, k1f, (short)0, s, false, false);
#pragma unroll
            for (int v = 0; v < 8; ++v) s_s[v + 8 * lh][j * 16 + lm] = s[v];
        }
        __syncthreads();

        // ---- streaming softmax on 16 rows (lanes 0..15); 0.125 = 1/sqrt(64) ----
        if (lane < 16) {
            const int r = lane, qrow_i = q0 + r;
            float sv[32];
            float bmax = -__builtin_inff();
#pragma unroll
            for (int c = 0; c < 32; ++c) {
                float v = (kb + c <= qrow_i) ? s_s[r][c] * 0.125f : -__builtin_inff();
                sv[c] = v; bmax = fmaxf(bmax, v);
            }
            float oldm = s_max[r];
            float nm = fmaxf(oldm, bmax);
            float scl = __expf(oldm - nm);
            float sum = 0.f;
#pragma unroll
            for (int c = 0; c < 32; ++c) {
                float p = (sv[c] == -__builtin_inff()) ? 0.f : __expf(sv[c] - nm);
                sum += p;
                s_p[r][c] = f2bf(p);
            }
            s_max[r] = nm;
            s_sum[r] = s_sum[r] * scl + sum;
            s_scale[r] = scl;
        }
        __syncthreads();

        // ---- rescale O accumulator ----
#pragma unroll
        for (int j = 0; j < 4; ++j)
#pragma unroll
            for (int v = 0; v < 8; ++v) o[j][v] *= s_scale[v + 8 * lh];

        // ---- P fragment (A layout, 16x32) from LDS ----
        v16bf pa;
#pragma unroll
        for (int e = 0; e < 8; ++e)  pa[e] = s_p[lm][lh * 16 + e];
#pragma unroll
        for (int e = 8; e < 16; ++e) pa[e] = s_p[lm][lh * 16 + e];

        // ---- O += P V : B-fragments are contiguous v16bf LDS loads ----
#pragma unroll
        for (int j = 0; j < 4; ++j) {
            v16bf vb = *(const v16bf*)(&s_vt[j * 16 + lm][lh * 16]);
            o[j] = __builtin_amdgcn_wmma_f32_16x16x32_bf16(false, pa, false, vb, (short)0, o[j], false, false);
        }
        __syncthreads();
    }

    // ---- normalize and store bf16 (swizzled A-tile for proj GEMM) ----
#pragma unroll
    for (int j = 0; j < 4; ++j)
#pragma unroll
        for (int v = 0; v < 8; ++v) {
            const int m = v + 8 * lh;
            y[aswz(b * TT + q0 + m, hh * HDIM + j * 16 + lm, CC)] = f2bf(o[j][v] / s_sum[m]);
        }
}

// ---------------------------------------------------------------------------
// Per-row cross-entropy via log-sum-exp over V; then global mean
// ---------------------------------------------------------------------------
__global__ __launch_bounds__(256)
void ce_kernel(const float* __restrict__ logits, const int* __restrict__ tgt,
               float* __restrict__ ce) {
    const int r = blockIdx.x, t = threadIdx.x;
    const float* lr = logits + (size_t)r * VV;
    __shared__ float red[256];
    float lmax = -__builtin_inff();
    for (int v = t; v < VV; v += 256) lmax = fmaxf(lmax, lr[v]);
    red[t] = lmax; __syncthreads();
    for (int s = 128; s > 0; s >>= 1) { if (t < s) red[t] = fmaxf(red[t], red[t + s]); __syncthreads(); }
    const float mx = red[0];
    __syncthreads();
    float lsum = 0.f;
    for (int v = t; v < VV; v += 256) lsum += __expf(lr[v] - mx);
    red[t] = lsum; __syncthreads();
    for (int s = 128; s > 0; s >>= 1) { if (t < s) red[t] += red[t + s]; __syncthreads(); }
    if (t == 0) {
        const float lse = mx + __logf(red[0]);
        const int tg = tgt[r];
        ce[r] = (tg != -1) ? (lse - lr[tg]) : 0.f;
    }
}

__global__ __launch_bounds__(256)
void loss_kernel(const float* __restrict__ ce, const int* __restrict__ tgt,
                 float* __restrict__ out) {
    __shared__ float rs[256], rc[256];
    const int t = threadIdx.x;
    float s = 0.f, c = 0.f;
    for (int r = t; r < BT; r += 256) { s += ce[r]; c += (tgt[r] != -1) ? 1.f : 0.f; }
    rs[t] = s; rc[t] = c; __syncthreads();
    for (int st = 128; st > 0; st >>= 1) {
        if (t < st) { rs[t] += rs[t + st]; rc[t] += rc[t + st]; }
        __syncthreads();
    }
    if (t == 0) out[0] = rs[0] / fmaxf(rc[0], 1.f);
}

// ---------------------------------------------------------------------------
extern "C" void kernel_launch(void* const* d_in, const int* in_sizes, int n_in,
                              void* d_out, int out_size, void* d_ws, size_t ws_size,
                              hipStream_t stream) {
    const float* wte    = (const float*)d_in[0];
    const float* wpe    = (const float*)d_in[1];
    const float* ln1_s  = (const float*)d_in[2];
    const float* ln1_b  = (const float*)d_in[3];
    const float* attn_w = (const float*)d_in[4];
    const float* attn_b = (const float*)d_in[5];
    const float* proj_w = (const float*)d_in[6];
    const float* proj_b = (const float*)d_in[7];
    const float* ln2_s  = (const float*)d_in[8];
    const float* ln2_b  = (const float*)d_in[9];
    const float* fc_w   = (const float*)d_in[10];
    const float* fc_b   = (const float*)d_in[11];
    const float* fcp_w  = (const float*)d_in[12];
    const float* fcp_b  = (const float*)d_in[13];
    const float* lnf_s  = (const float*)d_in[14];
    const float* lnf_b  = (const float*)d_in[15];
    const int*   idx    = (const int*)d_in[16];
    const int*   tgt    = (const int*)d_in[17];

    char* base = (char*)d_ws;
    float*  x    = (float*)base;   base += (size_t)BT * CC * 4;        // fp32 residual
    __bf16* h    = (__bf16*)base;  base += (size_t)BT * CC * 2;        // ln out (swizzled)
    __bf16* y    = (__bf16*)base;  base += (size_t)BT * CC * 2;        // attn out (swizzled)
    __bf16* qkvb = (__bf16*)base;  base += (size_t)BT * 3 * CC * 2;    // qkv (row-major)
    __bf16* fcb  = (__bf16*)base;  base += (size_t)BT * 4 * CC * 2;    // gelu(fc) (swizzled)
    __bf16* wb   = (__bf16*)base;  base += (size_t)VV * CC * 2;        // weight scratch
    float*  ce   = (float*)base;

    float* outp   = (float*)d_out;
    float* logits = outp + 1;                                          // [BT, V]

    const dim3 gblk(128);
    const dim3 ggrid512((CC + 63) / 64, BT / 128);
    const dim3 ggrid1536((3 * CC + 63) / 64, BT / 128);
    const dim3 ggrid2048((4 * CC + 63) / 64, BT / 128);
    const dim3 ggridV((VV + 63) / 64, BT / 128);

    {
        size_t n = (size_t)BT * CC;
        embed_kernel<<<dim3((unsigned)((n + 255) / 256)), 256, 0, stream>>>(wte, wpe, idx, x);
    }

    for (int l = 0; l < LL; ++l) {
        // ---- attention sublayer ----
        ln_kernel<<<BT, 256, 0, stream>>>(x, ln1_s + (size_t)l * CC, ln1_b + (size_t)l * CC, h);
        convtr_kernel<<<dim3(3 * CC / 32, CC / 32), 256, 0, stream>>>(
            attn_w + (size_t)l * CC * 3 * CC, wb, CC, 3 * CC);
        wmma_gemm<false, false, true, 1><<<ggrid1536, gblk, 0, stream>>>(
            h, wb, attn_b + (size_t)l * 3 * CC, nullptr, qkvb, BT, 3 * CC, CC);
        attn_kernel<<<dim3(TT / 16, HH, BB), 32, 0, stream>>>(qkvb, y);
        convtr_kernel<<<dim3(CC / 32, CC / 32), 256, 0, stream>>>(
            proj_w + (size_t)l * CC * CC, wb, CC, CC);
        wmma_gemm<false, true, true, 0><<<ggrid512, gblk, 0, stream>>>(
            y, wb, proj_b + (size_t)l * CC, x, x, BT, CC, CC);
        // ---- MLP sublayer ----
        ln_kernel<<<BT, 256, 0, stream>>>(x, ln2_s + (size_t)l * CC, ln2_b + (size_t)l * CC, h);
        convtr_kernel<<<dim3(4 * CC / 32, CC / 32), 256, 0, stream>>>(
            fc_w + (size_t)l * CC * 4 * CC, wb, CC, 4 * CC);
        wmma_gemm<true, false, true, 2><<<ggrid2048, gblk, 0, stream>>>(
            h, wb, fc_b + (size_t)l * 4 * CC, nullptr, fcb, BT, 4 * CC, CC);
        convtr_kernel<<<dim3(CC / 32, 4 * CC / 32), 256, 0, stream>>>(
            fcp_w + (size_t)l * 4 * CC * CC, wb, 4 * CC, CC);
        wmma_gemm<false, true, true, 0><<<ggrid512, gblk, 0, stream>>>(
            fcb, wb, fcp_b + (size_t)l * CC, x, x, BT, CC, 4 * CC);
    }

    // ---- final LN + tied LM head + loss ----
    ln_kernel<<<BT, 256, 0, stream>>>(x, lnf_s, lnf_b, h);
    conv_kernel<<<dim3((unsigned)(((size_t)VV * CC + 255) / 256)), 256, 0, stream>>>(
        wte, wb, (size_t)VV * CC);
    wmma_gemm<false, false, false, 0><<<ggridV, gblk, 0, stream>>>(
        h, wb, nullptr, nullptr, logits, BT, VV, CC);
    ce_kernel<<<BT, 256, 0, stream>>>(logits, tgt, ce);
    loss_kernel<<<1, 256, 0, stream>>>(ce, tgt, outp);
}
